// AttentionHead_13572096655785
// MI455X (gfx1250) — matrix operand order repaired and checked by reading, not compiled, not run on hardware
//
#include <hip/hip_runtime.h>
#include <hip/hip_bf16.h>

// ---------------------------------------------------------------------------
// Attention head, MI455X (gfx1250, wave32, WMMA bf16 -> f32 accumulate).
//   B=4, T=2048, C=1024, HEAD=2048
// Stage 1: convert x -> bf16, W -> bf16 transposed (for contiguous B-fragments)
// Stage 2: qkv projection GEMMs via v_wmma_f32_16x16x32_bf16
//          q,k row-major bf16; v stored transposed [B][HEAD][T]
// Stage 3: flash attention, online softmax, deterministic LDS reduction.
// Workspace usage: 124 MiB of d_ws.
// ---------------------------------------------------------------------------

typedef __bf16 bf16_t;
typedef __attribute__((ext_vector_type(16))) bf16_t v16bf;
typedef __attribute__((ext_vector_type(8)))  float  v8f;

#define BATCH 4
#define SEQ   2048
#define CDIM  1024
#define HEADD 2048
#define MROWS (BATCH * SEQ)   /* 8192 */
#define NEG_BIG (-3.0e38f)

union Frag16 {
    v16bf v;
    uint4 q[2];
    unsigned short h[16];
};
union Pack8 {
    uint4 q;
    unsigned short h[8];
};

static __device__ __forceinline__ unsigned short f32_to_bf16(float f) {
    unsigned int u = __float_as_uint(f);
    unsigned int r = u + 0x7FFFu + ((u >> 16) & 1u);   // round-to-nearest-even
    return (unsigned short)(r >> 16);
}

// ---------------------------------------------------------------------------
// Stage 1 kernels: fp32 -> bf16 conversion (elementwise and transposed)
// ---------------------------------------------------------------------------
__global__ void cvt_f32_to_bf16(const float* __restrict__ in,
                                unsigned short* __restrict__ out, int n) {
    int i = blockIdx.x * blockDim.x + threadIdx.x;
    if (i < n) out[i] = f32_to_bf16(in[i]);
}

// in: [C][H] fp32 -> out: [H][C] bf16
__global__ void cvt_transpose_bf16(const float* __restrict__ in,
                                   unsigned short* __restrict__ out,
                                   int C, int H) {
    int i = blockIdx.x * blockDim.x + threadIdx.x;
    if (i < C * H) {
        int h = i / C, c = i - h * C;
        out[i] = f32_to_bf16(in[c * H + h]);
    }
}

// ---------------------------------------------------------------------------
// Stage 2: Y = x*W + b. One wave computes one 16x16 tile, K-loop of 32 WMMAs.
//   Xb : [MROWS][CDIM]  bf16 (A, row-major)
//   WT : [HEADD][CDIM]  bf16 (B as W-transposed -> contiguous per-lane loads)
//   dst: transposed==0 -> [MROWS][HEADD] bf16 (q,k)
//        transposed==1 -> [BATCH][HEADD][SEQ] bf16 (vT)
// ---------------------------------------------------------------------------
__global__ __launch_bounds__(256)
void proj_gemm(const unsigned short* __restrict__ Xb,
               const unsigned short* __restrict__ WT,
               const float* __restrict__ bias,
               unsigned short* __restrict__ dst,
               int transposed) {
    const int lane  = threadIdx.x & 31;
    const int wave  = threadIdx.x >> 5;
    const int m0    = blockIdx.x * 16;
    const int n0    = (blockIdx.y * 8 + wave) * 16;
    const int col   = lane & 15;   // A row / B col / D col owned by this lane
    const int hi    = lane >> 4;   // lane half selects K sub-chunks
    const int half8 = hi * 8;

    const unsigned short* arow = Xb + (size_t)(m0 + col) * CDIM;
    const unsigned short* brow = WT + (size_t)(n0 + col) * CDIM + half8;

    v8f acc = {0.f, 0.f, 0.f, 0.f, 0.f, 0.f, 0.f, 0.f};
#pragma unroll 4
    for (int k0 = 0; k0 < CDIM; k0 += 32) {
        Frag16 a, bm;
        a.q[0]  = *(const uint4*)(arow + k0 + half8);        // K = k0 + half8 + 0..7
        a.q[1]  = *(const uint4*)(arow + k0 + 16 + half8);   // K = k0+16+half8 + 0..7
        bm.q[0] = *(const uint4*)(brow + k0);                // K = k0 + half8 + 0..7
        bm.q[1] = *(const uint4*)(brow + k0 + 16);
        acc = __builtin_amdgcn_wmma_f32_16x16x32_bf16(false, a.v, false, bm.v,
                                                      (short)0, acc, false, false);
    }
    const float bv = bias[n0 + col];
    if (!transposed) {
        // D layout: lane owns column (n0+col), rows m0 + hi*8 + r
#pragma unroll
        for (int r = 0; r < 8; ++r) {
            int m = m0 + half8 + r;
            dst[(size_t)m * HEADD + n0 + col] = f32_to_bf16(acc[r] + bv);
        }
    } else {
        // vT store: 8 consecutive t values per lane -> one 16B store
        int m = m0 + half8;
        int bb = m / SEQ, t = m - bb * SEQ;
        Pack8 p;
#pragma unroll
        for (int r = 0; r < 8; ++r) p.h[r] = f32_to_bf16(acc[r] + bv);
        *(uint4*)(dst + ((size_t)bb * HEADD + (n0 + col)) * SEQ + t) = p.q;
    }
}

// ---------------------------------------------------------------------------
// Stage 3: flash attention. Grid = (SEQ/16, BATCH), block = 256 (8 wave32).
// Per 32-key step:
//   - 8 waves split the 16x32 score tile by (key-half x 512-wide h range),
//     16 WMMAs each; partials reduced in LDS in a FIXED order (deterministic).
//   - every wave redundantly does online softmax for its lane's row, building
//     the P fragment directly in the 16x32 A-layout (row = lane%16).
//   - each wave updates its 16 private 16x16 output tiles with P * vT WMMAs.
// ---------------------------------------------------------------------------
__global__ __launch_bounds__(256)
void flash_attn(const unsigned short* __restrict__ Q,    // [MROWS][HEADD] bf16
                const unsigned short* __restrict__ K,    // [MROWS][HEADD] bf16
                const unsigned short* __restrict__ VT,   // [BATCH][HEADD][SEQ] bf16
                float* __restrict__ Out) {               // [BATCH][SEQ][HEADD] f32
    __shared__ float Spart[8][16][17];   // per-wave partial 16x16 score tiles
    __shared__ float Sacc[16][33];       // reduced 16x32 score tile (padded)
    __shared__ float AlphaL[16];
    __shared__ float InvL[16];

    const int tid   = threadIdx.x;
    const int lane  = tid & 31;
    const int wave  = tid >> 5;
    const int mt    = blockIdx.x;                 // 16-row tile in sequence
    const int b     = blockIdx.y;
    const int row0  = mt * 16;
    const int col   = lane & 15;
    const int hi    = lane >> 4;
    const int half8 = hi * 8;
    const int grow  = row0 + col;                 // this lane's global row (stats)

    const int hf    = wave & 1;                   // which 16-key half of 32-key block
    const int hbase = (wave >> 1) * (HEADD / 4);  // 512-wide h range

    const unsigned short* qrow  = Q  + (size_t)(b * SEQ + row0 + col) * HEADD;
    const unsigned short* kbase = K  + (size_t)b * SEQ * HEADD;
    const unsigned short* vbase = VT + (size_t)b * HEADD * SEQ;

    v8f acc[16];
    const v8f vzero = {0.f, 0.f, 0.f, 0.f, 0.f, 0.f, 0.f, 0.f};
#pragma unroll
    for (int i = 0; i < 16; ++i) acc[i] = vzero;

    float m_i = NEG_BIG;
    float l_i = 0.0f;
    const float scale = 0.03125f;                 // C^-0.5 = 1/sqrt(1024)

    const int nsteps = (16 * mt + 47) / 32;       // causal: keys 0 .. row0+15

    for (int st = 0; st < nsteps; ++st) {
        const int s_base = st * 32;

        // ---- partial scores: keys s_base + hf*16 .. +15 over h in [hbase, hbase+512)
        v8f sp = vzero;
        const unsigned short* krow =
            kbase + (size_t)(s_base + hf * 16 + col) * HEADD + hbase + half8;
#pragma unroll 4
        for (int hc = 0; hc < HEADD / 4; hc += 32) {
            Frag16 a, bk;
            a.q[0]  = *(const uint4*)(qrow + hbase + hc + half8);
            a.q[1]  = *(const uint4*)(qrow + hbase + hc + 16 + half8);
            bk.q[0] = *(const uint4*)(krow + hc);
            bk.q[1] = *(const uint4*)(krow + hc + 16);
            sp = __builtin_amdgcn_wmma_f32_16x16x32_bf16(false, a.v, false, bk.v,
                                                         (short)0, sp, false, false);
        }
#pragma unroll
        for (int r = 0; r < 8; ++r) Spart[wave][half8 + r][col] = sp[r];
        __syncthreads();

        // ---- deterministic 4-way reduction into Sacc (fixed summation order)
        {
            int o = tid;
#pragma unroll
            for (int rep = 0; rep < 2; ++rep, o += 256) {
                int m = o >> 5, j = o & 31;
                int jh = j >> 4, jc = j & 15;
                Sacc[m][j] = ((Spart[jh][m][jc] + Spart[jh + 2][m][jc]) +
                              (Spart[jh + 4][m][jc] + Spart[jh + 6][m][jc]));
            }
        }
        __syncthreads();

        // ---- online softmax for row (lane & 15); identical across waves/halves
        const int mrow = col;
        float mu = NEG_BIG;
#pragma unroll
        for (int j = 0; j < 32; ++j) {
            float s = (s_base + j <= grow) ? Sacc[mrow][j] * scale : NEG_BIG;
            mu = fmaxf(mu, s);
        }
        float m_new = fmaxf(m_i, mu);
        float sigma = 0.0f;
#pragma unroll
        for (int j = 0; j < 32; ++j) {
            float s = (s_base + j <= grow) ? Sacc[mrow][j] * scale : NEG_BIG;
            sigma += __expf(s - m_new);
        }
        float alpha = __expf(m_i - m_new);
        l_i = alpha * l_i + sigma;
        m_i = m_new;

        // ---- P fragment in the 16x32 A layout (element e -> key column j)
        Frag16 pf;
#pragma unroll
        for (int e = 0; e < 16; ++e) {
            int j = ((e >> 3) << 4) + half8 + (e & 7);
            float s = (s_base + j <= grow) ? Sacc[mrow][j] * scale : NEG_BIG;
            pf.h[e] = f32_to_bf16(__expf(s - m_new));
        }
        if (tid < 16) AlphaL[tid] = alpha;
        __syncthreads();

        // ---- rescale accumulators, then accumulate P * vT
        float arow[8];
#pragma unroll
        for (int r = 0; r < 8; ++r) arow[r] = AlphaL[half8 + r];
#pragma unroll
        for (int ct = 0; ct < 16; ++ct) {
#pragma unroll
            for (int r = 0; r < 8; ++r) acc[ct][r] *= arow[r];
        }
#pragma unroll
        for (int ct = 0; ct < 16; ++ct) {
            const int n0 = (wave * 16 + ct) * 16;
            const unsigned short* vrow =
                vbase + (size_t)(n0 + col) * SEQ + s_base + half8;
            Frag16 bvf;
            bvf.q[0] = *(const uint4*)(vrow);
            bvf.q[1] = *(const uint4*)(vrow + 16);
            acc[ct] = __builtin_amdgcn_wmma_f32_16x16x32_bf16(false, pf.v, false, bvf.v,
                                                              (short)0, acc[ct],
                                                              false, false);
        }
    }

    // ---- normalize by l and store fp32 output
    if (tid < 16) InvL[tid] = 1.0f / l_i;   // wave0 lanes 0..15 own rows 0..15
    __syncthreads();
    float inv[8];
#pragma unroll
    for (int r = 0; r < 8; ++r) inv[r] = InvL[half8 + r];
#pragma unroll
    for (int ct = 0; ct < 16; ++ct) {
        const int n0 = (wave * 16 + ct) * 16;
#pragma unroll
        for (int r = 0; r < 8; ++r) {
            int m = row0 + half8 + r;
            Out[(size_t)(b * SEQ + m) * HEADD + n0 + col] = acc[ct][r] * inv[r];
        }
    }
}

// ---------------------------------------------------------------------------
// Host-side launch
// ---------------------------------------------------------------------------
extern "C" void kernel_launch(void* const* d_in, const int* in_sizes, int n_in,
                              void* d_out, int out_size, void* d_ws, size_t ws_size,
                              hipStream_t stream) {
    (void)in_sizes; (void)n_in; (void)out_size; (void)ws_size;
    const float* x  = (const float*)d_in[0];
    const float* Wq = (const float*)d_in[1];
    const float* bq = (const float*)d_in[2];
    const float* Wk = (const float*)d_in[3];
    const float* bk = (const float*)d_in[4];
    const float* Wv = (const float*)d_in[5];
    const float* bv = (const float*)d_in[6];
    float* out = (float*)d_out;

    char* ws = (char*)d_ws;
    unsigned short* xb  = (unsigned short*)(ws);                          // 16 MiB
    unsigned short* wqT = (unsigned short*)(ws + ((size_t)16 << 20));     //  4 MiB
    unsigned short* wkT = (unsigned short*)(ws + ((size_t)20 << 20));     //  4 MiB
    unsigned short* wvT = (unsigned short*)(ws + ((size_t)24 << 20));     //  4 MiB
    unsigned short* q   = (unsigned short*)(ws + ((size_t)28 << 20));     // 32 MiB
    unsigned short* k   = (unsigned short*)(ws + ((size_t)60 << 20));     // 32 MiB
    unsigned short* vT  = (unsigned short*)(ws + ((size_t)92 << 20));     // 32 MiB (=124 MiB total)

    {   // x -> bf16
        int n = MROWS * CDIM;
        cvt_f32_to_bf16<<<(n + 255) / 256, 256, 0, stream>>>(x, xb, n);
    }
    {   // W -> bf16 transposed
        int n = CDIM * HEADD;
        int g = (n + 255) / 256;
        cvt_transpose_bf16<<<g, 256, 0, stream>>>(Wq, wqT, CDIM, HEADD);
        cvt_transpose_bf16<<<g, 256, 0, stream>>>(Wk, wkT, CDIM, HEADD);
        cvt_transpose_bf16<<<g, 256, 0, stream>>>(Wv, wvT, CDIM, HEADD);
    }
    dim3 pg(MROWS / 16, HEADD / (16 * 8));
    proj_gemm<<<pg, 256, 0, stream>>>(xb, wqT, bq, q, 0);
    proj_gemm<<<pg, 256, 0, stream>>>(xb, wkT, bk, k, 0);
    proj_gemm<<<pg, 256, 0, stream>>>(xb, wvT, bv, vT, 1);

    dim3 ag(SEQ / 16, BATCH);
    flash_attn<<<ag, 256, 0, stream>>>(q, k, vT, out);
}
